// MolEGNN_21208548508108
// MI455X (gfx1250) — compile-verified, matrix operand order checked by reading, not run on hardware
//
#include <hip/hip_runtime.h>
#include <hip/hip_bf16.h>

// ---------------- problem constants ----------------
#define N_NODES   100000
#define N_EDGES   1000000
#define N_GRAPHS  1024
#define NODE_DIM  39
#define EDGE_DIM  13
#define HID       64
#define LN_EPS    1e-5f

typedef _Float16 f16;
typedef __attribute__((ext_vector_type(16))) _Float16 v16h;
typedef __attribute__((ext_vector_type(8)))  _Float16 v8h;
typedef __attribute__((ext_vector_type(4)))  _Float16 v4h;
typedef __attribute__((ext_vector_type(8)))  float    v8f;
typedef __attribute__((ext_vector_type(4)))  float    v4f;

// wave-local LDS write->read ordering (DS ops are in-order per wave; wait drains them)
#define LDS_SYNC() asm volatile("s_wait_dscnt 0" ::: "memory")

#define WMMA_F16(a,b,c) __builtin_amdgcn_wmma_f32_16x16x32_f16(false,(a),false,(b),(short)0,(c),false,false)

// ---- fragment loads -------------------------------------------------------
// A fragment (16xK f16, ISA 7.12.2 layout) from an LDS tile, row stride KP halves.
__device__ __forceinline__ v16h frag_from_tile(const f16* tile, int kstep, int KP, int lane) {
  const int m    = lane & 15;
  const int koff = (lane & 16) ? 8 : 0;
  const f16* p = tile + m * KP + kstep * 32 + koff;
  v8h lo = *(const v8h*)p;
  v8h hi = *(const v8h*)(p + 16);
  v16h a;
#pragma unroll
  for (int i = 0; i < 8; ++i) { a[i] = lo[i]; a[i + 8] = hi[i]; }
  return a;
}

// B fragment from pre-packed weights: 16 contiguous halves per (frag,lane).
__device__ __forceinline__ v16h frag_from_pack(const f16* wp, int frag, int lane) {
  const f16* p = wp + (size_t)(frag * 32 + lane) * 16;
  v8h lo = *(const v8h*)p;
  v8h hi = *(const v8h*)(p + 8);
  v16h b;
#pragma unroll
  for (int i = 0; i < 8; ++i) { b[i] = lo[i]; b[i + 8] = hi[i]; }
  return b;
}

// ---- weight packing: f32 (K x 64) -> f16 fragments matching B layout ------
__global__ void pack_w(const float* __restrict__ W, f16* __restrict__ out, int Kreal, int KP) {
  int t = blockIdx.x * blockDim.x + threadIdx.x;
  int nfrag = (KP / 32) * 4;
  if (t >= nfrag * 32) return;
  int frag = t >> 5, lane = t & 31;
  int kstep = frag >> 2, ntile = frag & 3;
  int n  = ntile * 16 + (lane & 15);
  int kb = kstep * 32 + ((lane & 16) ? 8 : 0);
  f16* o = out + (size_t)t * 16;
#pragma unroll
  for (int j = 0; j < 16; ++j) {
    int k = kb + ((j < 8) ? j : (j + 8));
    float v = (k < Kreal) ? W[(size_t)k * 64 + n] : 0.f;
    o[j] = (f16)v;
  }
}

// ---- utility kernels ------------------------------------------------------
__global__ void zero_f32(float* p, int n) {
  for (int i = blockIdx.x * blockDim.x + threadIdx.x; i < n; i += gridDim.x * blockDim.x)
    p[i] = 0.f;
}

__global__ void count_kernel(const int* __restrict__ dst, float* __restrict__ cnt, int E) {
  int i = blockIdx.x * blockDim.x + threadIdx.x;
  if (i < E) atomicAdd(&cnt[dst[i]], 1.f);
}

// ---- edge message kernel: one wave per 16-edge tile -----------------------
// mi = [x[dst] | x[src] | e] (K padded to KP1); h = relu(mi@W1+b1); m = h@W2+b2;
// m *= sigmoid(e@Gw+Gb); atomicAdd into agg[dst].
template <int IN, int KP1>
__global__ void edge_kernel(const float* __restrict__ x32, const f16* __restrict__ h16,
                            const int* __restrict__ src, const int* __restrict__ dst,
                            const float* __restrict__ ea,
                            const f16* __restrict__ w1p, const float* __restrict__ b1,
                            const f16* __restrict__ w2p, const float* __restrict__ b2,
                            const f16* __restrict__ gwp, const float* __restrict__ gb,
                            float* __restrict__ agg, int ntiles) {
  constexpr int WAVES = 4;
  __shared__ __align__(16) f16 At[WAVES][16 * KP1];
  __shared__ __align__(16) f16 Et[WAVES][16 * 32];
  __shared__ __align__(16) f16 Ht[WAVES][16 * 64];
  const int wave = threadIdx.x >> 5, lane = threadIdx.x & 31;
  const int tile = blockIdx.x * WAVES + wave;
  if (tile >= ntiles) return;
  const int e0 = tile * 16;
  f16* A = At[wave]; f16* Eb = Et[wave]; f16* H = Ht[wave];

  if (e0 + 16 < N_EDGES) __builtin_prefetch(ea + (size_t)(e0 + 16) * EDGE_DIM, 0, 1);

  if constexpr (IN == HID) {
    // Vectorized gather: each node row is 64 f16 = 8 chunks of 16B.
    // 16 rows x (8 dst chunks + 8 src chunks) = 256 chunks; b128 load -> b128 DS store.
    for (int ch = lane; ch < 256; ch += 32) {
      int r  = ch >> 4;
      int cc = ch & 15;               // 0..7: dst row, 8..15: src row
      int e  = e0 + r;
      int node = (cc < 8) ? dst[e] : src[e];
      int k8 = (cc & 7) * 8;
      v8h val = *(const v8h*)(h16 + (size_t)node * HID + k8);
      *(v8h*)(A + r * KP1 + ((cc < 8) ? 0 : HID) + k8) = val;
    }
    // edge-attr columns (128..140) + zero pad (141..159); mirror into gate tile
    for (int idx = lane; idx < 16 * 32; idx += 32) {
      int r = idx >> 5, k = idx & 31;
      float v = (k < EDGE_DIM) ? ea[(size_t)(e0 + r) * EDGE_DIM + k] : 0.f;
      A[r * KP1 + 2 * HID + k] = (f16)v;
      Eb[idx] = (f16)v;
    }
  } else {
    // layer 0: 39-float rows, scalar staging
    for (int idx = lane; idx < 16 * KP1; idx += 32) {
      int r = idx / KP1, k = idx - r * KP1;
      int e = e0 + r;
      float v = 0.f;
      if (k < IN)              v = x32[(size_t)dst[e] * NODE_DIM + k];
      else if (k < 2 * IN)     v = x32[(size_t)src[e] * NODE_DIM + (k - IN)];
      else if (k < 2 * IN + EDGE_DIM) v = ea[(size_t)e * EDGE_DIM + (k - 2 * IN)];
      A[idx] = (f16)v;
    }
    for (int idx = lane; idx < 16 * 32; idx += 32) {
      int r = idx >> 5, k = idx & 31;
      Eb[idx] = (f16)((k < EDGE_DIM) ? ea[(size_t)(e0 + r) * EDGE_DIM + k] : 0.f);
    }
  }
  LDS_SYNC();

  const int c = lane & 15, rh = (lane >> 4) * 8;

  // hidden = relu(mi @ W1 + b1) -> LDS (f16)
#pragma unroll
  for (int t = 0; t < 4; ++t) {
    v8f acc = {};
#pragma unroll
    for (int ks = 0; ks < KP1 / 32; ++ks) {
      v16h a = frag_from_tile(A, ks, KP1, lane);
      v16h b = frag_from_pack(w1p, ks * 4 + t, lane);
      acc = WMMA_F16(a, b, acc);
    }
    float bias = b1[t * 16 + c];
#pragma unroll
    for (int v = 0; v < 8; ++v) {
      float hv = acc[v] + bias;
      H[(rh + v) * 64 + t * 16 + c] = (f16)(hv > 0.f ? hv : 0.f);
    }
  }
  LDS_SYNC();

  // m = hidden @ W2 + b2; gated by sigmoid(e @ Gw + Gb); scatter-add
#pragma unroll
  for (int t = 0; t < 4; ++t) {
    v8f acc = {};
#pragma unroll
    for (int ks = 0; ks < 2; ++ks) {
      v16h a = frag_from_tile(H, ks, 64, lane);
      v16h b = frag_from_pack(w2p, ks * 4 + t, lane);
      acc = WMMA_F16(a, b, acc);
    }
    v8f gacc = {};
    {
      v16h a = frag_from_tile(Eb, 0, 32, lane);
      v16h b = frag_from_pack(gwp, t, lane);
      gacc = WMMA_F16(a, b, gacc);
    }
    float bias = b2[t * 16 + c], gbias = gb[t * 16 + c];
#pragma unroll
    for (int v = 0; v < 8; ++v) {
      float g = gacc[v] + gbias;
      float s = 1.f / (1.f + __expf(-g));
      float m = (acc[v] + bias) * s;
      int e = e0 + rh + v;
      atomicAdd(&agg[(size_t)dst[e] * HID + t * 16 + c], m);
    }
  }
}

// ---- node update kernel: one wave per 16-node tile ------------------------
// cat = [x | agg/cnt] (KP=128); out = relu(cat@W1+b1)@W2+b2; +res; LN; relu.
template <int IN>
__global__ void node_kernel(const float* __restrict__ x32, const f16* __restrict__ hin,
                            const float* __restrict__ agg, const float* __restrict__ cnt,
                            const f16* __restrict__ w1p, const float* __restrict__ b1,
                            const f16* __restrict__ w2p, const float* __restrict__ b2,
                            const f16* __restrict__ rwp, const float* __restrict__ rb,
                            const float* __restrict__ lng, const float* __restrict__ lnb,
                            f16* __restrict__ hout, int ntiles) {
  constexpr int KP = 128, WAVES = 4;
  __shared__ __align__(16) f16   At[WAVES][16 * KP];
  __shared__ __align__(16) f16   Ht[WAVES][16 * 64];
  __shared__ __align__(16) float St[WAVES][16 * 64];
  const int wave = threadIdx.x >> 5, lane = threadIdx.x & 31;
  const int tile = blockIdx.x * WAVES + wave;
  if (tile >= ntiles) return;
  const int n0 = tile * 16;
  f16* A = At[wave]; f16* H = Ht[wave]; float* S = St[wave];

  if constexpr (IN == HID) {
    // x part: 16 rows x 8 chunks of 16B, contiguous
    for (int ch = lane; ch < 128; ch += 32) {
      int r = ch >> 3, k8 = (ch & 7) * 8;
      *(v8h*)(A + r * KP + k8) = *(const v8h*)(hin + (size_t)(n0 + r) * HID + k8);
    }
    // agg part: 16 rows x 16 chunks of float4, divide by degree, convert f16
    for (int ch = lane; ch < 256; ch += 32) {
      int r = ch >> 4, k4 = (ch & 15) * 4;
      int node = n0 + r;
      float cc = cnt[node]; cc = cc > 1.f ? cc : 1.f;
      float rinv = 1.f / cc;
      v4f av = *(const v4f*)(agg + (size_t)node * HID + k4);
      v4h o;
#pragma unroll
      for (int j = 0; j < 4; ++j) o[j] = (f16)(av[j] * rinv);
      *(v4h*)(A + r * KP + HID + k4) = o;
    }
  } else {
    for (int idx = lane; idx < 16 * KP; idx += 32) {
      int r = idx >> 7, k = idx & 127;
      int node = n0 + r;
      float v = 0.f;
      if (k < IN) {
        v = x32[(size_t)node * NODE_DIM + k];
      } else if (k < IN + HID) {
        float cc = cnt[node]; cc = cc > 1.f ? cc : 1.f;
        v = agg[(size_t)node * HID + (k - IN)] / cc;
      }
      A[idx] = (f16)v;
    }
  }
  LDS_SYNC();

  const int c = lane & 15, rh = (lane >> 4) * 8;

#pragma unroll
  for (int t = 0; t < 4; ++t) {
    v8f acc = {};
#pragma unroll
    for (int ks = 0; ks < 4; ++ks) {
      v16h a = frag_from_tile(A, ks, KP, lane);
      v16h b = frag_from_pack(w1p, ks * 4 + t, lane);
      acc = WMMA_F16(a, b, acc);
    }
    float bias = b1[t * 16 + c];
#pragma unroll
    for (int v = 0; v < 8; ++v) {
      float hv = acc[v] + bias;
      H[(rh + v) * 64 + t * 16 + c] = (f16)(hv > 0.f ? hv : 0.f);
    }
  }
  LDS_SYNC();

#pragma unroll
  for (int t = 0; t < 4; ++t) {
    v8f acc = {};
#pragma unroll
    for (int ks = 0; ks < 2; ++ks) {
      v16h a = frag_from_tile(H, ks, 64, lane);
      v16h b = frag_from_pack(w2p, ks * 4 + t, lane);
      acc = WMMA_F16(a, b, acc);
    }
    v8f racc = {};
    if constexpr (IN == NODE_DIM) {
      // res = x @ res_w + res_b ; res_w packed with Kreal=39 (zeros above)
#pragma unroll
      for (int ks = 0; ks < 2; ++ks) {
        v16h a = frag_from_tile(A, ks, KP, lane);
        v16h b = frag_from_pack(rwp, ks * 4 + t, lane);
        racc = WMMA_F16(a, b, racc);
      }
    }
    float bias = b2[t * 16 + c];
    float rbias = (IN == NODE_DIM) ? rb[t * 16 + c] : 0.f;
#pragma unroll
    for (int v = 0; v < 8; ++v) {
      float res;
      if constexpr (IN == NODE_DIM) res = racc[v] + rbias;
      else                          res = (float)A[(rh + v) * KP + t * 16 + c];
      S[(rh + v) * 64 + t * 16 + c] = acc[v] + bias + res;
    }
  }
  LDS_SYNC();

  // LayerNorm + relu per row; lanes 0..15 each own one row
  if (lane < 16) {
    const float* row = &S[lane * 64];
    float mu = 0.f;
#pragma unroll
    for (int f = 0; f < 64; ++f) mu += row[f];
    mu *= (1.f / 64.f);
    float var = 0.f;
#pragma unroll
    for (int f = 0; f < 64; ++f) { float d = row[f] - mu; var += d * d; }
    var *= (1.f / 64.f);
    float rs = rsqrtf(var + LN_EPS);
    f16* o = hout + (size_t)(n0 + lane) * HID;
#pragma unroll
    for (int f = 0; f < 64; ++f) {
      float v = (row[f] - mu) * rs * lng[f] + lnb[f];
      o[f] = (f16)(v > 0.f ? v : 0.f);
    }
  }
}

// ---- global mean pool -----------------------------------------------------
__global__ void pool_kernel(const f16* __restrict__ h, const int* __restrict__ batch,
                            float* __restrict__ pool, float* __restrict__ pcnt) {
  int i = blockIdx.x * blockDim.x + threadIdx.x;
  if (i >= N_NODES) return;
  int g = batch[i];
  const f16* hr = h + (size_t)i * HID;
#pragma unroll 8
  for (int f = 0; f < HID; ++f) atomicAdd(&pool[(size_t)g * HID + f], (float)hr[f]);
  atomicAdd(&pcnt[g], 1.f);
}

// ---- readout: one wave per graph ------------------------------------------
__global__ void readout_kernel(const float* __restrict__ pool, const float* __restrict__ pcnt,
                               const float* __restrict__ w1, const float* __restrict__ b1,
                               const float* __restrict__ w2, const float* __restrict__ b2,
                               float* __restrict__ out) {
  int wave = threadIdx.x >> 5, lane = threadIdx.x & 31;
  int g = blockIdx.x * (blockDim.x >> 5) + wave;
  if (g >= N_GRAPHS) return;
  float cc = pcnt[g]; cc = cc > 1.f ? cc : 1.f;
  const float* gp = pool + (size_t)g * HID;
  float d0 = b1[lane], d1 = b1[lane + 32];
#pragma unroll 8
  for (int k = 0; k < HID; ++k) {
    float gv = gp[k] / cc;
    d0 += gv * w1[(size_t)k * 64 + lane];
    d1 += gv * w1[(size_t)k * 64 + lane + 32];
  }
  d0 = d0 > 0.f ? d0 : 0.f;
  d1 = d1 > 0.f ? d1 : 0.f;
  float partial = d0 * w2[lane] + d1 * w2[lane + 32];
#pragma unroll
  for (int off = 16; off > 0; off >>= 1) partial += __shfl_down(partial, off, 32);
  if (lane == 0) out[g] = partial + b2[0];
}

// ---------------------------------------------------------------------------
extern "C" void kernel_launch(void* const* d_in, const int* in_sizes, int n_in,
                              void* d_out, int out_size, void* d_ws, size_t ws_size,
                              hipStream_t stream) {
  (void)in_sizes; (void)n_in; (void)out_size; (void)ws_size;
  const float* x    = (const float*)d_in[0];
  const int*   eidx = (const int*)d_in[1];
  const int*   srcI = eidx;             // edge_index[0]
  const int*   dstI = eidx + N_EDGES;   // edge_index[1]
  const float* ea   = (const float*)d_in[2];
  const int*   batch= (const int*)d_in[3];

  // param leaves in jax pytree order (dict keys sorted; list order for layers)
  auto F = [&](int i) { return (const float*)d_in[i]; };
  const float *gate_b[3], *gate_w[3], *ln_b[3], *ln_g[3], *msg_b1[3], *msg_b2[3],
              *msg_w1[3], *msg_w2[3], *node_b1[3], *node_b2[3], *node_w1[3], *node_w2[3];
  const float *res_b = nullptr, *res_w = nullptr;
  int idx = 4;
  for (int l = 0; l < 3; ++l) {
    gate_b[l] = F(idx++); gate_w[l] = F(idx++); ln_b[l] = F(idx++); ln_g[l] = F(idx++);
    msg_b1[l] = F(idx++); msg_b2[l] = F(idx++); msg_w1[l] = F(idx++); msg_w2[l] = F(idx++);
    node_b1[l]= F(idx++); node_b2[l]= F(idx++); node_w1[l]= F(idx++); node_w2[l]= F(idx++);
    if (l == 0) { res_b = F(idx++); res_w = F(idx++); }
  }
  const float* ro_b1 = F(idx++); const float* ro_b2 = F(idx++);
  const float* ro_w1 = F(idx++); const float* ro_w2 = F(idx++);

  // ---- workspace bump allocator ----
  char* base = (char*)d_ws; size_t off = 0;
  auto alloc = [&](size_t bytes) -> void* {
    void* p = base + off; off = (off + bytes + 255) & ~(size_t)255; return p;
  };
  const int KP1_L[3] = {96, 160, 160};   // msg_w1 K pads
  f16 *pk_mw1[3], *pk_mw2[3], *pk_gw[3], *pk_nw1[3], *pk_nw2[3], *pk_rw;
  for (int l = 0; l < 3; ++l) {
    pk_mw1[l] = (f16*)alloc((size_t)(KP1_L[l] / 32) * 4 * 512 * 2);
    pk_mw2[l] = (f16*)alloc((size_t)2 * 4 * 512 * 2);
    pk_gw[l]  = (f16*)alloc((size_t)1 * 4 * 512 * 2);
    pk_nw1[l] = (f16*)alloc((size_t)4 * 4 * 512 * 2);
    pk_nw2[l] = (f16*)alloc((size_t)2 * 4 * 512 * 2);
  }
  pk_rw = (f16*)alloc((size_t)2 * 4 * 512 * 2);
  float* agg  = (float*)alloc((size_t)N_NODES * HID * 4);
  float* cnt  = (float*)alloc((size_t)N_NODES * 4);
  f16*   hA   = (f16*)alloc((size_t)N_NODES * HID * 2);
  f16*   hB   = (f16*)alloc((size_t)N_NODES * HID * 2);
  float* pool = (float*)alloc((size_t)N_GRAPHS * HID * 4);
  float* pcnt = (float*)alloc((size_t)N_GRAPHS * 4);

  auto pack = [&](const float* W, f16* dst, int Kreal, int KP) {
    int threads = (KP / 32) * 4 * 32;
    pack_w<<<(threads + 255) / 256, 256, 0, stream>>>(W, dst, Kreal, KP);
  };
  auto zero = [&](float* p, int n) {
    int blocks = (n + 255) / 256; if (blocks > 1024) blocks = 1024;
    zero_f32<<<blocks, 256, 0, stream>>>(p, n);
  };

  // ---- pack all weights (f32 -> WMMA f16 fragments) ----
  const int MW1_K[3] = {2 * NODE_DIM + EDGE_DIM, 2 * HID + EDGE_DIM, 2 * HID + EDGE_DIM}; // 91,141,141
  const int NW1_K[3] = {NODE_DIM + HID, 2 * HID, 2 * HID};                                 // 103,128,128
  for (int l = 0; l < 3; ++l) {
    pack(msg_w1[l],  pk_mw1[l], MW1_K[l], KP1_L[l]);
    pack(msg_w2[l],  pk_mw2[l], HID, 64);
    pack(gate_w[l],  pk_gw[l],  EDGE_DIM, 32);
    pack(node_w1[l], pk_nw1[l], NW1_K[l], 128);
    pack(node_w2[l], pk_nw2[l], HID, 64);
  }
  pack(res_w, pk_rw, NODE_DIM, 64);

  // ---- in-degree counts (same for all layers) ----
  zero(cnt, N_NODES);
  count_kernel<<<(N_EDGES + 255) / 256, 256, 0, stream>>>(dstI, cnt, N_EDGES);

  const int ETILES = N_EDGES / 16;          // 62500
  const int NTILES = N_NODES / 16;          // 6250
  const int EBLK = ETILES / 4;              // 15625 (exact)
  const int NBLK = (NTILES + 3) / 4;        // 1563

  // ---- layer 0 (in = x, f32, dim 39) ----
  zero(agg, N_NODES * HID);
  edge_kernel<NODE_DIM, 96><<<EBLK, 128, 0, stream>>>(
      x, nullptr, srcI, dstI, ea,
      pk_mw1[0], msg_b1[0], pk_mw2[0], msg_b2[0], pk_gw[0], gate_b[0], agg, ETILES);
  node_kernel<NODE_DIM><<<NBLK, 128, 0, stream>>>(
      x, nullptr, agg, cnt,
      pk_nw1[0], node_b1[0], pk_nw2[0], node_b2[0], pk_rw, res_b,
      ln_g[0], ln_b[0], hA, NTILES);

  // ---- layers 1,2 (in = h, f16, dim 64) ----
  f16* hin = hA; f16* hout = hB;
  for (int l = 1; l < 3; ++l) {
    zero(agg, N_NODES * HID);
    edge_kernel<HID, 160><<<EBLK, 128, 0, stream>>>(
        nullptr, hin, srcI, dstI, ea,
        pk_mw1[l], msg_b1[l], pk_mw2[l], msg_b2[l], pk_gw[l], gate_b[l], agg, ETILES);
    node_kernel<HID><<<NBLK, 128, 0, stream>>>(
        nullptr, hin, agg, cnt,
        pk_nw1[l], node_b1[l], pk_nw2[l], node_b2[l], nullptr, nullptr,
        ln_g[l], ln_b[l], hout, NTILES);
    f16* t = hin; hin = hout; hout = t;
  }
  // final h is in `hin` after swap

  // ---- global mean pool + readout ----
  zero(pool, N_GRAPHS * HID);
  zero(pcnt, N_GRAPHS);
  pool_kernel<<<(N_NODES + 255) / 256, 256, 0, stream>>>(hin, batch, pool, pcnt);
  readout_kernel<<<(N_GRAPHS + 7) / 8, 256, 0, stream>>>(
      pool, pcnt, ro_w1, ro_b1, ro_w2, ro_b2, (float*)d_out);
}
